// GNNEncoder_43190191129144
// MI455X (gfx1250) — matrix operand order repaired and checked by reading, not compiled, not run on hardware
//
#include <hip/hip_runtime.h>
#include <hip/hip_bf16.h>

#define HID 128
#define HEADS 4
#define CH 32

typedef __attribute__((ext_vector_type(8)))  float  v8f;
typedef __attribute__((ext_vector_type(16))) __bf16 bf16x16;

// ---------------- small device helpers ----------------
__device__ __forceinline__ float4 ld4(const float* p) { return *(const float4*)p; }
__device__ __forceinline__ void   st4(float* p, float4 v) { *(float4*)p = v; }

__device__ __forceinline__ float geluf(float x) {
    return 0.5f * x * (1.0f + erff(x * 0.70710678118654752440f));
}
__device__ __forceinline__ float lrelu(float x) { return x > 0.f ? x : 0.2f * x; }

// order-preserving float<->uint encoding so atomicMax(uint) == float max.
// enc(x) > 0 for every finite x and -inf, so memset(0) is a valid -inf init.
__device__ __forceinline__ unsigned enc_f(float f) {
    unsigned u = __float_as_uint(f);
    return (u & 0x80000000u) ? ~u : (u | 0x80000000u);
}
__device__ __forceinline__ float dec_f(unsigned u) {
    return __uint_as_float((u & 0x80000000u) ? (u & 0x7FFFFFFFu) : ~u);
}

__device__ __forceinline__ unsigned pack_bf16(float a, float b) {
    union { __bf16 h[2]; unsigned u; } p;
    p.h[0] = (__bf16)a; p.h[1] = (__bf16)b;
    return p.u;
}
__device__ __forceinline__ int imin(int a, int b) { return a < b ? a : b; }

// ---------------- GEMM: C[M x 128] = A[M x K] @ W[K x 128] + bias ----------------
// block = 256 threads = 8 waves; 32-row M stripe per block; each wave owns one
// 16-col N tile and does TWO v_wmma_f32_16x16x32_bf16 per K-step (reusing its
// B fragment for both 16-row A subtiles).  All staging loads are branch-free:
// clamped addresses + cndmask-to-zero, so loads batch under one wait.
#define AS_STRIDE 34   // +2 bf16 pad -> 17-bank row stride, conflict free
#define BS_STRIDE 34   // stored [n][k]

__global__ void gemm_bias_128(const float* __restrict__ A, const float* __restrict__ W,
                              const float* __restrict__ bias, float* __restrict__ C,
                              int M, int K)
{
    __shared__ unsigned lds_raw[544 + 2176];   // As: 32*34 bf16, Bs: 128*34 bf16
    __bf16* As = (__bf16*)lds_raw;
    __bf16* Bs = (__bf16*)(lds_raw + 544);

    const int t    = threadIdx.x;
    const int w    = t >> 5;
    const int lane = t & 31;
    const int m0   = blockIdx.x * 32;

    v8f acc0 = {0.f, 0.f, 0.f, 0.f, 0.f, 0.f, 0.f, 0.f};
    v8f acc1 = {0.f, 0.f, 0.f, 0.f, 0.f, 0.f, 0.f, 0.f};

    const int KT = (K + 31) / 32;
    for (int kt = 0; kt < KT; ++kt) {
        const int kk = kt * 32;
        // ---- A tile: 32 rows x 32 k -> 512 bf16 pairs, 2 per thread ----
#pragma unroll
        for (int p = 0; p < 2; ++p) {
            int idx  = t + 256 * p;            // 0..511
            int row  = idx >> 4;
            int k    = (idx & 15) << 1;
            int grow = m0 + row;
            int ga   = imin(grow, M - 1);      // clamped, always in-bounds
            int gk   = imin(kk + k, K - 2);    // K is even -> pair stays aligned
            float2 av = *(const float2*)&A[(size_t)ga * K + gk];
            bool valid = (grow < M) && (kk + k < K);
            *(unsigned*)&As[row * AS_STRIDE + k] = valid ? pack_bf16(av.x, av.y) : 0u;
        }
        // ---- B tile: 32 k x 128 n via float4 (b128) loads, scatter to [n][k] ----
#pragma unroll
        for (int g = 0; g < 4; ++g) {
            int fv = t + 256 * g;              // 0..1023 float4 chunks
            int k  = fv >> 5;
            int n0 = (fv & 31) << 2;
            int gk = imin(kk + k, K - 1);
            float4 wv = ld4(W + (size_t)gk * 128 + n0);
            bool valid = (kk + k) < K;
            Bs[(n0 + 0) * BS_STRIDE + k] = (__bf16)(valid ? wv.x : 0.f);
            Bs[(n0 + 1) * BS_STRIDE + k] = (__bf16)(valid ? wv.y : 0.f);
            Bs[(n0 + 2) * BS_STRIDE + k] = (__bf16)(valid ? wv.z : 0.f);
            Bs[(n0 + 3) * BS_STRIDE + k] = (__bf16)(valid ? wv.w : 0.f);
        }
        __syncthreads();
        // ---- fragments per documented CDNA5 16-bit layouts ----
        union { bf16x16 v; unsigned u[8]; } ufa0, ufa1, ufb;
#pragma unroll
        for (int e2 = 0; e2 < 8; ++e2) {
            int ka = ((e2 >> 2) << 4) + ((lane >> 4) << 3) + ((e2 & 3) << 1); // A K-halves
            int kb = ((e2 >> 2) << 4) + ((lane >> 4) << 3) + ((e2 & 3) << 1); // B K-halves
            ufa0.u[e2] = *(const unsigned*)&As[(lane & 15) * AS_STRIDE + ka];
            ufa1.u[e2] = *(const unsigned*)&As[(16 + (lane & 15)) * AS_STRIDE + ka];
            ufb.u[e2]  = *(const unsigned*)&Bs[(w * 16 + (lane & 15)) * BS_STRIDE + kb];
        }
        acc0 = __builtin_amdgcn_wmma_f32_16x16x32_bf16(false, ufa0.v, false, ufb.v,
                                                       (short)0, acc0, false, false);
        acc1 = __builtin_amdgcn_wmma_f32_16x16x32_bf16(false, ufa1.v, false, ufb.v,
                                                       (short)0, acc1, false, false);
        __syncthreads();
    }
    // ---- store D (+bias): lane -> col = 16w + L%16, VGPR v -> row v + 8*(L>=16) ----
    const int col = w * 16 + (lane & 15);
    const float bv = bias[col];
    const int rbase = m0 + ((lane >> 4) << 3);
    float* __restrict__ Cp = C + (size_t)rbase * 128 + col;
    if (m0 + 32 <= M) {
        // fast path: whole 32-row stripe in range, branch-free stores
#pragma unroll
        for (int v = 0; v < 8; ++v) {
            Cp[(size_t)v * 128]        = acc0[v] + bv;
            Cp[(size_t)(v + 16) * 128] = acc1[v] + bv;
        }
    } else {
#pragma unroll
        for (int v = 0; v < 8; ++v) {
            if (rbase + v < M)      Cp[(size_t)v * 128]        = acc0[v] + bv;
            if (rbase + v + 16 < M) Cp[(size_t)(v + 16) * 128] = acc1[v] + bv;
        }
    }
}

// ---------------- LN(x)*g+b then GELU, wave per row, in place ----------------
__global__ void ln_gelu_kernel(float* __restrict__ x, const float* __restrict__ g,
                               const float* __restrict__ b, int n)
{
    int row = blockIdx.x * 8 + (threadIdx.x >> 5);
    if (row >= n) return;
    int lane = threadIdx.x & 31, c0 = lane << 2;
    size_t off = (size_t)row * 128 + c0;
    float4 t = ld4(x + off);
    float s = t.x + t.y + t.z + t.w;
    for (int m = 1; m < 32; m <<= 1) s += __shfl_xor(s, m, 32);
    float mean = s * (1.f / 128.f);
    float dx = t.x - mean, dy = t.y - mean, dz = t.z - mean, dw = t.w - mean;
    float q = dx * dx + dy * dy + dz * dz + dw * dw;
    for (int m = 1; m < 32; m <<= 1) q += __shfl_xor(q, m, 32);
    float r = rsqrtf(q * (1.f / 128.f) + 1e-5f);
    float4 gv = ld4(g + c0), bv = ld4(b + c0);
    float4 y;
    y.x = geluf(dx * r * gv.x + bv.x);
    y.y = geluf(dy * r * gv.y + bv.y);
    y.z = geluf(dz * r * gv.z + bv.z);
    y.w = geluf(dw * r * gv.w + bv.w);
    st4(x + off, y);
}

// ---------------- x += pos_emb[time[row]] ----------------
__global__ void pos_add_kernel(float* __restrict__ x, const int* __restrict__ tm,
                               const float* __restrict__ pos, int n)
{
    int idx = blockIdx.x * blockDim.x + threadIdx.x;
    if (idx >= n * 32) return;
    int row = idx >> 5, c0 = (idx & 31) << 2;
    int tt = tm[row];
    float4 p = ld4(pos + (size_t)tt * 128 + c0);
    size_t off = (size_t)row * 128 + c0;
    float4 v = ld4(x + off);
    v.x += p.x; v.y += p.y; v.z += p.z; v.w += p.w;
    st4(x + off, v);
}

// ---------------- mean of ea columns (for self-loop fill) ----------------
__global__ void mean_ea_kernel(const float* __restrict__ ea, int E, int dim,
                               float* __restrict__ out)
{
    int t = blockIdx.x * blockDim.x + threadIdx.x;
    int stride = gridDim.x * blockDim.x;
    float s0 = 0.f, s1 = 0.f;
    for (int e = t; e < E; e += stride) {
        s0 += ea[(size_t)e * dim];
        if (dim == 2) s1 += ea[(size_t)e * dim + 1];
    }
    float inv = 1.f / (float)E;
    atomicAdd(&out[0], s0 * inv);
    if (dim == 2) atomicAdd(&out[1], s1 * inv);
}

// ---------------- GATv2 pass 1: edge score + segment max (wave per edge) ----------------
__global__ void gat_score_kernel(const float* __restrict__ xl, const float* __restrict__ xr,
                                 const int* __restrict__ ei, const float* __restrict__ ea,
                                 int edge_dim, const float* __restrict__ We,
                                 const float* __restrict__ att, const float* __restrict__ mean_ea,
                                 float* __restrict__ score, unsigned* __restrict__ segmax,
                                 int E_real, int n_total)
{
    int e = blockIdx.x * 8 + (threadIdx.x >> 5);
    if (e >= n_total) return;
    int lane = threadIdx.x & 31;
    int src, dst; float e0, e1 = 0.f;
    if (e < E_real) {
        src = ei[e]; dst = ei[E_real + e];
        e0 = ea[(size_t)e * edge_dim];
        if (edge_dim == 2) e1 = ea[(size_t)e * edge_dim + 1];
    } else {                      // virtual self loop, edge_attr = mean(ea)
        src = dst = e - E_real;
        e0 = mean_ea[0];
        if (edge_dim == 2) e1 = mean_ea[1];
    }
    int c0 = lane << 2;
    int h  = lane >> 3;
    float4 a  = ld4(xl + (size_t)src * 128 + c0);
    float4 b  = ld4(xr + (size_t)dst * 128 + c0);
    float4 w0 = ld4(We + c0);
    float eex = e0 * w0.x, eey = e0 * w0.y, eez = e0 * w0.z, eew = e0 * w0.w;
    if (edge_dim == 2) {
        float4 w1 = ld4(We + 128 + c0);
        eex += e1 * w1.x; eey += e1 * w1.y; eez += e1 * w1.z; eew += e1 * w1.w;
    }
    float zx = lrelu(a.x + b.x + eex);
    float zy = lrelu(a.y + b.y + eey);
    float zz = lrelu(a.z + b.z + eez);
    float zw = lrelu(a.w + b.w + eew);
    float4 at = ld4(att + h * 32 + (c0 & 31));
    float s = zx * at.x + zy * at.y + zz * at.z + zw * at.w;
    s += __shfl_xor(s, 1, 8);
    s += __shfl_xor(s, 2, 8);
    s += __shfl_xor(s, 4, 8);
    if ((lane & 7) == 0) {
        score[(size_t)e * 4 + h] = s;
        atomicMax(&segmax[(size_t)dst * 4 + h], enc_f(s));
    }
}

// ---------------- pass 2: exp(score - max), segment sum ----------------
__global__ void gat_norm_kernel(const int* __restrict__ ei, float* __restrict__ score,
                                const unsigned* __restrict__ segmax, float* __restrict__ segz,
                                int E_real, int n_total)
{
    int idx = blockIdx.x * blockDim.x + threadIdx.x;
    if (idx >= n_total * 4) return;
    int e = idx >> 2, h = idx & 3;
    int dst = (e < E_real) ? ei[E_real + e] : (e - E_real);
    float mx = dec_f(segmax[(size_t)dst * 4 + h]);
    float ex = __expf(score[idx] - mx);
    score[idx] = ex;
    atomicAdd(&segz[(size_t)dst * 4 + h], ex);
}

// ---------------- pass 3: out[dst] += alpha * xl[src] ----------------
__global__ void gat_scatter_kernel(const float* __restrict__ xl, const int* __restrict__ ei,
                                   const float* __restrict__ score, const float* __restrict__ segz,
                                   float* __restrict__ out, int E_real, int n_total)
{
    int idx = blockIdx.x * blockDim.x + threadIdx.x;
    if (idx >= n_total * 32) return;
    int e = idx >> 5, c4 = idx & 31, c0 = c4 << 2, h = c4 >> 3;
    int src, dst;
    if (e < E_real) { src = ei[e]; dst = ei[E_real + e]; }
    else            { src = dst = e - E_real; }
    float alpha = score[(size_t)e * 4 + h] / (segz[(size_t)dst * 4 + h] + 1e-16f);
    float4 v = ld4(xl + (size_t)src * 128 + c0);
    float* o = out + (size_t)dst * 128 + c0;
    atomicAdd(o + 0, alpha * v.x);
    atomicAdd(o + 1, alpha * v.y);
    atomicAdd(o + 2, alpha * v.z);
    atomicAdd(o + 3, alpha * v.w);
}

// ---------------- x = LN(gelu(o + b1 + b2) + x) * g + b  (wave per row) ----------------
__global__ void layer_post_kernel(const float* __restrict__ o, const float* __restrict__ b1,
                                  const float* __restrict__ b2, float* __restrict__ x,
                                  const float* __restrict__ g, const float* __restrict__ b, int n)
{
    int row = blockIdx.x * 8 + (threadIdx.x >> 5);
    if (row >= n) return;
    int lane = threadIdx.x & 31, c0 = lane << 2;
    size_t off = (size_t)row * 128 + c0;
    float4 ov = ld4(o + off), b1v = ld4(b1 + c0), b2v = ld4(b2 + c0), xv = ld4(x + off);
    float tx = geluf(ov.x + b1v.x + b2v.x) + xv.x;
    float ty = geluf(ov.y + b1v.y + b2v.y) + xv.y;
    float tz = geluf(ov.z + b1v.z + b2v.z) + xv.z;
    float tw = geluf(ov.w + b1v.w + b2v.w) + xv.w;
    float s = tx + ty + tz + tw;
    for (int m = 1; m < 32; m <<= 1) s += __shfl_xor(s, m, 32);
    float mean = s * (1.f / 128.f);
    float dx = tx - mean, dy = ty - mean, dz = tz - mean, dw = tw - mean;
    float q = dx * dx + dy * dy + dz * dz + dw * dw;
    for (int m = 1; m < 32; m <<= 1) q += __shfl_xor(q, m, 32);
    float r = rsqrtf(q * (1.f / 128.f) + 1e-5f);
    float4 gv = ld4(g + c0), bv = ld4(b + c0);
    float4 y;
    y.x = dx * r * gv.x + bv.x;
    y.y = dy * r * gv.y + bv.y;
    y.z = dz * r * gv.z + bv.z;
    y.w = dw * r * gv.w + bv.w;
    st4(x + off, y);
}

// ---------------- host orchestration ----------------
extern "C" void kernel_launch(void* const* d_in, const int* in_sizes, int n_in,
                              void* d_out, int out_size, void* d_ws, size_t ws_size,
                              hipStream_t stream)
{
    (void)n_in; (void)out_size; (void)ws_size;

    const float* x_wifi = (const float*)d_in[0];
    const float* x_imu  = (const float*)d_in[1];
    const int*   t_wifi = (const int*)d_in[2];
    const int*   t_imu  = (const int*)d_in[3];
    const int*   ei_ww  = (const int*)d_in[4];
    const float* ea_ww  = (const float*)d_in[5];
    const int*   ei_ii  = (const int*)d_in[6];
    const float* ea_ii  = (const float*)d_in[7];
    const int*   ei_iw  = (const int*)d_in[8];
    const float* ea_iw  = (const float*)d_in[9];
    const int*   ei_wi  = (const int*)d_in[10];
    const float* ea_wi  = (const float*)d_in[11];

    const int NW = in_sizes[2];
    const int NI = in_sizes[3];
    const int E  = in_sizes[5];       // ea_ww is (E,1)
    const int KW = in_sizes[0] / NW;  // 120
    const int KI = in_sizes[1] / NI;  // 60
    const int NMAX = NW > NI ? NW : NI;

    // params pytree flattened with sorted dict keys after the 12 arrays
    const float* pp[96];
    for (int i = 12; i < n_in && i < 96; ++i) pp[i] = (const float*)d_in[i];
    // imu_proj: W1 W2 b1 b2 be1 g1
    const float* imuW1 = pp[12], *imuW2 = pp[13], *imub1 = pp[14], *imub2 = pp[15],
               * imube1 = pp[16], *imug1 = pp[17];
    const float* pos_emb = pp[82];
    const float* wifW1 = pp[83], *wifW2 = pp[84], *wifb1 = pp[85], *wifb2 = pp[86],
               * wifbe1 = pp[87], *wifg1 = pp[88];

    // workspace layout (f32)
    float* w = (float*)d_ws;
    float*    xw     = w; w += (size_t)NW * 128;
    float*    xi     = w; w += (size_t)NI * 128;
    float*    bufA   = w; w += (size_t)NMAX * 128;   // xl
    float*    bufB   = w; w += (size_t)NMAX * 128;   // xr
    float*    ow     = w; w += (size_t)NW * 128;
    float*    oi     = w; w += (size_t)NI * 128;
    float*    score  = w; w += (size_t)(E + NMAX) * 4;
    unsigned* segmax = (unsigned*)w; w += (size_t)NMAX * 4;
    float*    segz   = w; w += (size_t)NMAX * 4;
    float*    meanb  = w; w += 4;

    const dim3 B256(256);
    auto blks = [](long n) { return dim3((unsigned)((n + 255) / 256)); };
    auto gblk = [](int m) { return dim3((unsigned)((m + 31) / 32)); };

    // ---- input MLPs + positional embedding ----
    gemm_bias_128<<<gblk(NW), B256, 0, stream>>>(x_wifi, wifW1, wifb1, ow, NW, KW);
    ln_gelu_kernel<<<dim3((NW + 7) / 8), B256, 0, stream>>>(ow, wifg1, wifbe1, NW);
    gemm_bias_128<<<gblk(NW), B256, 0, stream>>>(ow, wifW2, wifb2, xw, NW, 128);
    pos_add_kernel<<<blks((long)NW * 32), B256, 0, stream>>>(xw, t_wifi, pos_emb, NW);

    gemm_bias_128<<<gblk(NI), B256, 0, stream>>>(x_imu, imuW1, imub1, oi, NI, KI);
    ln_gelu_kernel<<<dim3((NI + 7) / 8), B256, 0, stream>>>(oi, imug1, imube1, NI);
    gemm_bias_128<<<gblk(NI), B256, 0, stream>>>(oi, imuW2, imub2, xi, NI, 128);
    pos_add_kernel<<<blks((long)NI * 32), B256, 0, stream>>>(xi, t_imu, pos_emb, NI);

    // one GATv2 conv, accumulating atomically into out_accum
    auto run_conv = [&](const float* xs, int n_src, const float* xd, int n_dst,
                        const int* ei, const float* ea, int edge_dim, bool self_loops,
                        const float* const* P, float* out_accum) {
        const float* We = P[0], *Wl = P[1], *Wr = P[2], *att = P[3], *bl = P[5], *br = P[6];
        int n_total = E + (self_loops ? n_dst : 0);
        hipMemsetAsync(segmax, 0, (size_t)n_dst * 4 * sizeof(unsigned), stream); // 0 < enc(-inf), acts as -inf
        hipMemsetAsync(segz,   0, (size_t)n_dst * 4 * sizeof(float), stream);
        if (self_loops) {
            hipMemsetAsync(meanb, 0, 2 * sizeof(float), stream);
            mean_ea_kernel<<<dim3(256), B256, 0, stream>>>(ea, E, edge_dim, meanb);
        }
        gemm_bias_128<<<dim3((n_src + 31) / 32), B256, 0, stream>>>(xs, Wl, bl, bufA, n_src, 128);
        gemm_bias_128<<<dim3((n_dst + 31) / 32), B256, 0, stream>>>(xd, Wr, br, bufB, n_dst, 128);
        gat_score_kernel<<<dim3((n_total + 7) / 8), B256, 0, stream>>>(
            bufA, bufB, ei, ea, edge_dim, We, att, meanb, score, segmax, E, n_total);
        gat_norm_kernel<<<blks((long)n_total * 4), B256, 0, stream>>>(
            ei, score, segmax, segz, E, n_total);
        gat_scatter_kernel<<<blks((long)n_total * 32), B256, 0, stream>>>(
            bufA, ei, score, segz, out_accum, E, n_total);
    };

    // ---- 2 GAT layers ----
    for (int l = 0; l < 2; ++l) {
        const float* const* L = &pp[18 + 32 * l];
        // leaf offsets inside a layer (sorted keys):
        //  ii:+0..6  iw:+7..13  ln_i_b:+14 ln_i_g:+15 ln_w_b:+16 ln_w_g:+17  wi:+18..24  ww:+25..31
        hipMemsetAsync(ow, 0, (size_t)NW * 128 * sizeof(float), stream);
        run_conv(xw, NW, xw, NW, ei_ww, ea_ww, 1, true,  L + 25, ow); // ww
        run_conv(xi, NI, xw, NW, ei_iw, ea_iw, 2, false, L + 7,  ow); // iw
        hipMemsetAsync(oi, 0, (size_t)NI * 128 * sizeof(float), stream);
        run_conv(xi, NI, xi, NI, ei_ii, ea_ii, 1, true,  L + 0,  oi); // ii
        run_conv(xw, NW, xi, NI, ei_wi, ea_wi, 1, false, L + 18, oi); // wi
        layer_post_kernel<<<dim3((NW + 7) / 8), B256, 0, stream>>>(
            ow, L[25 + 4], L[7 + 4],  xw, L[17], L[16], NW);
        layer_post_kernel<<<dim3((NI + 7) / 8), B256, 0, stream>>>(
            oi, L[0 + 4],  L[18 + 4], xi, L[15], L[14], NI);
    }

    // ---- outputs: concat(xw, xi) ----
    hipMemcpyAsync(d_out, xw, (size_t)NW * 128 * sizeof(float),
                   hipMemcpyDeviceToDevice, stream);
    hipMemcpyAsync((float*)d_out + (size_t)NW * 128, xi, (size_t)NI * 128 * sizeof(float),
                   hipMemcpyDeviceToDevice, stream);
}
